// FlowEmbedding_18588618457256
// MI455X (gfx1250) — matrix operand order repaired and checked by reading, not compiled, not run on hardware
//
#include <hip/hip_runtime.h>

typedef __bf16 bf16;
typedef __bf16 v16bf __attribute__((ext_vector_type(16)));
typedef __bf16 v8bf  __attribute__((ext_vector_type(8)));
typedef float  v8f   __attribute__((ext_vector_type(8)));
typedef int    v4i   __attribute__((ext_vector_type(4)));

constexpr int   BATCH  = 4;
constexpr int   NPTS   = 8192;   // 1 << 13
constexpr int   CFEAT  = 128;
constexpr int   CIN    = 256;
constexpr int   COUT   = 256;
constexpr int   KNN    = 16;
constexpr float BN_EPS = 1e-3f;

// LDS row stride for the staged weight panel: pad to kill bank conflicts
constexpr int WPAD = CIN + 8;    // 264 bf16 = 528 B -> lane l starts at bank 4l mod 64

#if defined(__has_builtin)
#if __has_builtin(__builtin_amdgcn_global_load_async_to_lds_b128) && \
    __has_builtin(__builtin_amdgcn_s_wait_asynccnt)
#define USE_ASYNC_LDS 1
#endif
#endif
#ifndef USE_ASYNC_LDS
#define USE_ASYNC_LDS 0
#endif

#define AS_GLOBAL __attribute__((address_space(1)))
#define AS_LDS    __attribute__((address_space(3)))

// ---------------------------------------------------------------------------
// Swizzled activation layout (contiguous-k B-operand layout used by this GEMM):
//   Xs[((b*NPTS + n) * 16 + kg) * 16 + e]   with  k = kg*16 + e
// A lane's B fragment (16 consecutive K at k0+16*half for its column) is
// 32 contiguous bytes.
// ---------------------------------------------------------------------------
__device__ __forceinline__ size_t xs_index(int b, int n, int kg) {
    return ((size_t)(b * NPTS + n) * 16 + kg) * 16;
}

// ---------------------------------------------------------------------------
// Kernel 1: kNN (k=16) + inverse-distance interpolation + concat,
// writes swizzled bf16 activations with packed 32-byte stores.
// ---------------------------------------------------------------------------
__global__ __launch_bounds__(256) void knn_interp_kernel(
    const float* __restrict__ p1, const float* __restrict__ p2,
    const float* __restrict__ f1, const float* __restrict__ f2,
    bf16* __restrict__ xs)
{
    __shared__ float sx[256], sy[256], sz[256];
    const int tid = threadIdx.x;
    const int b   = blockIdx.x / (NPTS / 256);
    const int n2  = (blockIdx.x % (NPTS / 256)) * 256 + tid;

    const float* p2b = p2 + (size_t)b * 3 * NPTS;
    const float qx = p2b[0 * NPTS + n2];
    const float qy = p2b[1 * NPTS + n2];
    const float qz = p2b[2 * NPTS + n2];

    float bd[KNN];
    int   bi[KNN];
#pragma unroll
    for (int i = 0; i < KNN; ++i) { bd[i] = 3.4e38f; bi[i] = 0; }

    const float* p1b = p1 + (size_t)b * 3 * NPTS;
    for (int t = 0; t < NPTS; t += 256) {
        __syncthreads();
        sx[tid] = p1b[0 * NPTS + t + tid];
        sy[tid] = p1b[1 * NPTS + t + tid];
        sz[tid] = p1b[2 * NPTS + t + tid];
        __syncthreads();
#pragma unroll 4
        for (int j = 0; j < 256; ++j) {
            const float dx = qx - sx[j];
            const float dy = qy - sy[j];
            const float dz = qz - sz[j];
            const float d  = dx * dx + dy * dy + dz * dz;
            if (d < bd[KNN - 1]) {
                bd[KNN - 1] = d;
                bi[KNN - 1] = t + j;
#pragma unroll
                for (int i = KNN - 1; i > 0; --i) {
                    if (bd[i] < bd[i - 1]) {
                        float td = bd[i]; bd[i] = bd[i - 1]; bd[i - 1] = td;
                        int   ti = bi[i]; bi[i] = bi[i - 1]; bi[i - 1] = ti;
                    }
                }
            }
        }
    }

    float w[KNN];
    float s = 0.0f;
#pragma unroll
    for (int i = 0; i < KNN; ++i) {
        const float dk = sqrtf(fmaxf(bd[i], 0.0f));
        w[i] = 1.0f / (dk + 1e-10f);
        s += w[i];
    }
    const float inv_s = 1.0f / s;
#pragma unroll
    for (int i = 0; i < KNN; ++i) w[i] *= inv_s;

    const float* f1b = f1 + (size_t)b * CFEAT * NPTS;
    const float* f2b = f2 + (size_t)b * CFEAT * NPTS;
    bf16* xcol = xs + xs_index(b, n2, 0);   // 256 elements for this column

    // interpolated features -> channels 0..127 (kg 0..7)
    for (int cg = 0; cg < CFEAT / 16; ++cg) {
        v16bf o;
#pragma unroll
        for (int e = 0; e < 16; ++e) {
            float acc = 0.0f;
#pragma unroll
            for (int i = 0; i < KNN; ++i)
                acc += w[i] * f1b[(size_t)(cg * 16 + e) * NPTS + bi[i]];
            o[e] = (bf16)acc;
        }
        *(v16bf*)(xcol + cg * 16) = o;
    }
    // features2 copy -> channels 128..255 (kg 8..15)
    for (int cg = 0; cg < CFEAT / 16; ++cg) {
        v16bf o;
#pragma unroll
        for (int e = 0; e < 16; ++e)
            o[e] = (bf16)f2b[(size_t)(cg * 16 + e) * NPTS + n2];
        *(v16bf*)(xcol + (CFEAT / 16 + cg) * 16) = o;
    }
}

// ---------------------------------------------------------------------------
// f32 -> bf16 conversion (weights, row major kept)
// ---------------------------------------------------------------------------
__global__ void cvt_f32_bf16_kernel(const float* __restrict__ in,
                                    bf16* __restrict__ out, int n)
{
    int i = blockIdx.x * blockDim.x + threadIdx.x;
    const int stride = gridDim.x * blockDim.x;
    for (; i < n; i += stride) out[i] = (bf16)in[i];
}

// ---------------------------------------------------------------------------
// Kernel 2: GEMM  Y[b][o][n] = sum_c W[o][c] * X[b][c][n] + bias[o]
// All 8 waves of a block share one 16-row M panel of W, staged in LDS
// via the gfx1250 async global->LDS path. Waves split 512 N columns.
// B fragments are contiguous 32-B loads from the swizzled activations.
// ---------------------------------------------------------------------------
__global__ __launch_bounds__(256) void gemm_bias_kernel(
    const bf16* __restrict__ Wb,   // [COUT][CIN] row major bf16
    const bf16* __restrict__ Xs,   // swizzled activations
    const float* __restrict__ bias,
    float* __restrict__ Y)         // [B][COUT][N] f32
{
    __shared__ __align__(32) bf16 sW[16 * WPAD];   // ~8.4 KB

    const int tid  = threadIdx.x;
    const int wv   = tid >> 5;
    const int lane = tid & 31;
    const int half = lane >> 4;
    const int l    = lane & 15;

    const int b     = blockIdx.x >> 8;            // / (16 mTiles * 16 nSupers)
    const int rem   = blockIdx.x & 255;
    const int mBase = (rem >> 4) << 4;
    const int nBase = (rem & 15) * 512 + wv * 64;

    // ---- stage the 16x256 weight panel into LDS ----
    {
        const int r  = tid >> 4;          // row 0..15
        const int ck = (tid & 15) * 16;   // 16-element chunk
        const bf16* src = Wb + (size_t)(mBase + r) * CIN + ck;
        bf16*       dst = &sW[r * WPAD + ck];
#if USE_ASYNC_LDS
        __builtin_amdgcn_global_load_async_to_lds_b128(
            (AS_GLOBAL v4i*)src, (AS_LDS v4i*)dst, 0, 0);
        __builtin_amdgcn_global_load_async_to_lds_b128(
            (AS_GLOBAL v4i*)(src + 8), (AS_LDS v4i*)(dst + 8), 0, 0);
        __builtin_amdgcn_s_wait_asynccnt(0);
#else
        v8bf t0 = *(const v8bf*)src;
        v8bf t1 = *(const v8bf*)(src + 8);
        *(v8bf*)dst       = t0;
        *(v8bf*)(dst + 8) = t1;
#endif
    }
    __syncthreads();

    v8f acc[4];
#pragma unroll
    for (int j = 0; j < 4; ++j)
#pragma unroll
        for (int i = 0; i < 8; ++i)
            acc[j][i] = bias[mBase + i + 8 * half];

#pragma unroll
    for (int k0 = 0; k0 < CIN; k0 += 32) {
        // A fragment from LDS: {k0+16h..+15} of row l
        v8bf alo = *(const v8bf*)&sW[l * WPAD + k0 + half * 16];
        v8bf ahi = *(const v8bf*)&sW[l * WPAD + k0 + half * 16 + 8];
        v16bf a = __builtin_shufflevector(alo, ahi,
                     0, 1, 2, 3, 4, 5, 6, 7, 8, 9, 10, 11, 12, 13, 14, 15);
        const int kg = (k0 >> 4) + half;
#pragma unroll
        for (int j = 0; j < 4; ++j) {
            const bf16* bp = Xs + xs_index(b, nBase + j * 16 + l, kg);
            v8bf blo = *(const v8bf*)bp;
            v8bf bhi = *(const v8bf*)(bp + 8);
            v16bf bv = __builtin_shufflevector(blo, bhi,
                         0, 1, 2, 3, 4, 5, 6, 7, 8, 9, 10, 11, 12, 13, 14, 15);
            acc[j] = __builtin_amdgcn_wmma_f32_16x16x32_bf16(
                false, a, false, bv, (short)0, acc[j], false, false);
        }
    }

    float* ybase = Y + (size_t)b * COUT * NPTS + (size_t)(mBase + 8 * half) * NPTS;
#pragma unroll
    for (int j = 0; j < 4; ++j) {
        const int col = nBase + j * 16 + l;
#pragma unroll
        for (int i = 0; i < 8; ++i)
            ybase[(size_t)i * NPTS + col] = acc[j][i];
    }
}

// ---------------------------------------------------------------------------
// Kernel 3: per-channel sum / sumsq over (B, N) — one block per channel
// ---------------------------------------------------------------------------
__global__ __launch_bounds__(256) void bn_stats_kernel(
    const float* __restrict__ Y, float* __restrict__ stats /* [2*COUT] */)
{
    __shared__ float ssum[256], ssq[256];
    const int o = blockIdx.x;
    float s = 0.0f, q = 0.0f;
    for (int i = threadIdx.x; i < BATCH * NPTS; i += 256) {
        const int bb = i >> 13;
        const int n  = i & (NPTS - 1);
        const float v = Y[(size_t)bb * COUT * NPTS + (size_t)o * NPTS + n];
        s += v; q += v * v;
    }
    ssum[threadIdx.x] = s; ssq[threadIdx.x] = q;
    __syncthreads();
    for (int st = 128; st > 0; st >>= 1) {
        if (threadIdx.x < st) {
            ssum[threadIdx.x] += ssum[threadIdx.x + st];
            ssq[threadIdx.x]  += ssq[threadIdx.x + st];
        }
        __syncthreads();
    }
    if (threadIdx.x == 0) { stats[o] = ssum[0]; stats[COUT + o] = ssq[0]; }
}

// ---------------------------------------------------------------------------
// Kernel 4a: BN + ReLU, writes swizzled bf16 (feeds next GEMM).
// One thread = one (b, n, kg) -> one packed 32-byte store.
// ---------------------------------------------------------------------------
__global__ __launch_bounds__(256) void bn_relu_swz_kernel(
    const float* __restrict__ Y, const float* __restrict__ stats,
    const float* __restrict__ g, const float* __restrict__ be,
    bf16* __restrict__ Xs)
{
    __shared__ float ssc[COUT], ssh[COUT];
    const int tid = threadIdx.x;
    {
        const float invM = 1.0f / (float)(BATCH * NPTS);
        const float mean = stats[tid] * invM;
        const float var  = stats[COUT + tid] * invM - mean * mean;
        const float sc   = g[tid] * rsqrtf(var + BN_EPS);
        ssc[tid] = sc;
        ssh[tid] = be[tid] - mean * sc;
    }
    __syncthreads();

    const int u  = blockIdx.x * 256 + tid;   // grid*block == B*NPTS*16
    const int kg = u & 15;
    const int n  = (u >> 4) & (NPTS - 1);
    const int b  = u >> 17;

    const float* yb = Y + ((size_t)b * COUT + kg * 16) * NPTS + n;
    v16bf o;
#pragma unroll
    for (int e = 0; e < 16; ++e) {
        const int c = kg * 16 + e;
        const float v = fmaxf(yb[(size_t)e * NPTS] * ssc[c] + ssh[c], 0.0f);
        o[e] = (bf16)v;
    }
    *(v16bf*)(Xs + xs_index(b, n, kg)) = o;
}

// ---------------------------------------------------------------------------
// Kernel 4b: BN + ReLU, final f32 output (linear layout)
// ---------------------------------------------------------------------------
__global__ __launch_bounds__(256) void bn_relu_out_kernel(
    const float* __restrict__ Y, const float* __restrict__ stats,
    const float* __restrict__ g, const float* __restrict__ be,
    float* __restrict__ out)
{
    __shared__ float ssc[COUT], ssh[COUT];
    const int tid = threadIdx.x;
    {
        const float invM = 1.0f / (float)(BATCH * NPTS);
        const float mean = stats[tid] * invM;
        const float var  = stats[COUT + tid] * invM - mean * mean;
        const float sc   = g[tid] * rsqrtf(var + BN_EPS);
        ssc[tid] = sc;
        ssh[tid] = be[tid] - mean * sc;
    }
    __syncthreads();

    const size_t total = (size_t)BATCH * COUT * NPTS;
    for (size_t i = blockIdx.x * 256ull + tid; i < total;
         i += (size_t)gridDim.x * 256ull) {
        const int c = (int)((i >> 13) & (COUT - 1));
        out[i] = fmaxf(Y[i] * ssc[c] + ssh[c], 0.0f);
    }
}

// ---------------------------------------------------------------------------
extern "C" void kernel_launch(void* const* d_in, const int* in_sizes, int n_in,
                              void* d_out, int out_size, void* d_ws, size_t ws_size,
                              hipStream_t stream)
{
    (void)in_sizes; (void)n_in; (void)out_size; (void)ws_size;
    const float* points1   = (const float*)d_in[0];
    const float* points2   = (const float*)d_in[1];
    const float* features1 = (const float*)d_in[2];
    const float* features2 = (const float*)d_in[3];
    const float* w0  = (const float*)d_in[4];
    const float* b0  = (const float*)d_in[5];
    const float* g0  = (const float*)d_in[6];
    const float* be0 = (const float*)d_in[7];
    const float* w1  = (const float*)d_in[8];
    const float* b1  = (const float*)d_in[9];
    const float* g1  = (const float*)d_in[10];
    const float* be1 = (const float*)d_in[11];
    // d_in[12] = k, fixed at 16 (compile-time KNN)

    char* ws = (char*)d_ws;
    // Layout: Xs swizzled bf16 (16MB) | Y f32 (32MB) | stats | w0b | w1b
    bf16*  xs     = (bf16*)ws;
    float* ybuf   = (float*)(ws + (size_t)16 * 1024 * 1024);
    float* stats0 = (float*)(ws + (size_t)48 * 1024 * 1024);
    float* stats1 = stats0 + 2 * COUT;
    bf16*  w0b    = (bf16*)(stats1 + 2 * COUT);
    bf16*  w1b    = w0b + COUT * CIN;

    // 1) kNN + interpolation + concat -> swizzled bf16 activations
    knn_interp_kernel<<<BATCH * (NPTS / 256), 256, 0, stream>>>(
        points1, points2, features1, features2, xs);

    // 2) weights to bf16
    cvt_f32_bf16_kernel<<<64, 256, 0, stream>>>(w0, w0b, COUT * CIN);
    cvt_f32_bf16_kernel<<<64, 256, 0, stream>>>(w1, w1b, COUT * CIN);

    const int gemmBlocks = BATCH * (COUT / 16) * (NPTS / 512);  // 1024
    const int swzBlocks  = (BATCH * NPTS * 16) / 256;           // 2048

    // 3) layer 0
    gemm_bias_kernel<<<gemmBlocks, 256, 0, stream>>>(w0b, xs, b0, ybuf);
    bn_stats_kernel<<<COUT, 256, 0, stream>>>(ybuf, stats0);
    bn_relu_swz_kernel<<<swzBlocks, 256, 0, stream>>>(ybuf, stats0, g0, be0, xs);

    // 4) layer 1
    gemm_bias_kernel<<<gemmBlocks, 256, 0, stream>>>(w1b, xs, b1, ybuf);
    bn_stats_kernel<<<COUT, 256, 0, stream>>>(ybuf, stats1);
    bn_relu_out_kernel<<<4096, 256, 0, stream>>>(ybuf, stats1, g1, be1,
                                                 (float*)d_out);
}
